// ConvLSTMRecurrentLayer_10247791968622
// MI455X (gfx1250) — compile-verified, matrix operand order checked by reading, not run on hardware
//
#include <hip/hip_runtime.h>
#include <cmath>

// ---------------------------------------------------------------------------
// ConvLSTM (B=8,T=12,H=W=96,C=F=32) as implicit-GEMM bf16 WMMA on gfx1250.
//   z = conv3x3([h+x ; h], [Wx ; Wh]) + b   (Cin=64, Cout=128)
//   gates i,f,c,o -> c' = f*c + i*tanh(zc); h' = o*tanh(c')
// h state ping-pongs through d_out time slices; c state in d_ws.
// ---------------------------------------------------------------------------

typedef __attribute__((ext_vector_type(16))) __bf16 bf16x16;
typedef __attribute__((ext_vector_type(8)))  __bf16 bf16x8;
typedef __attribute__((ext_vector_type(8)))  float  f32x8;
typedef __attribute__((ext_vector_type(4)))  unsigned short u16x4;

#define H_   96
#define W_   96
#define Cin  32
#define Fo   32
#define T_   12
#define NB   8
#define TILE 16
#define PR   18      // patch rows/cols with halo
#define CH   64      // combined input channels (h+x | h)
#define KC   18      // K chunks: 9 taps * (64/32)
#define NT   8       // N tiles: 128 / 16

union ABfrag { bf16x16 v; bf16x8 h[2]; };

__device__ __forceinline__ unsigned short f2bf(float f) {
    __bf16 b = (__bf16)f;
    return __builtin_bit_cast(unsigned short, b);
}
// clamp(0.2z+0.5, 0, 1): fma + med3
__device__ __forceinline__ float hsig(float z) {
    float y = __builtin_fmaf(0.2f, z, 0.5f);
    return fminf(fmaxf(y, 0.0f), 1.0f);
}
// CDNA5 has hardware V_TANH_F32 (TRANS32 class, co-executes with VALU).
__device__ __forceinline__ float fast_tanh(float x) {
#if __has_builtin(__builtin_amdgcn_tanhf)
    return __builtin_amdgcn_tanhf(x);
#else
    float r;
    asm volatile("v_tanh_f32 %0, %1" : "=v"(r) : "v"(x));
    return r;
#endif
}

// ---------------------------------------------------------------------------
// Pre-swizzle Wx(3,3,32,128) + Wh(3,3,32,128) f32 -> bf16 B-fragment stream:
//   wfrag[((kc*NT + n)*32 + lane)*16 + e]
// Element e of lane (g = lane>>4) holds K = (e<8 ? e : e+8) + 8*g within the
// 32-wide k-chunk (matches the documented 16-bit WMMA operand VGPR layout),
// column N = lane&15 of N-tile n, kin = (kc&1)*32 + K, tap = kc>>1.
// kin<32 -> Wx (applied to h+x), kin>=32 -> Wh (applied to h).
// ---------------------------------------------------------------------------
__global__ void prep_weights_kernel(const float* __restrict__ Wx,
                                    const float* __restrict__ Wh,
                                    unsigned short* __restrict__ wfrag) {
    int idx = blockIdx.x * blockDim.x + threadIdx.x;
    if (idx >= KC * NT * 32 * 16) return;
    int e    = idx & 15;
    int lane = (idx >> 4) & 31;
    int n    = (idx >> 9) & 7;
    int kc   = idx >> 12;
    int g    = lane >> 4;
    int kl   = (e < 8 ? e : e + 8) + (g << 3);       // K within 32-chunk
    int kin  = ((kc & 1) << 5) + kl;                 // 0..63 combined channel
    int tap  = kc >> 1;                              // 0..8  (dy*3+dx)
    int cout = (n << 4) + (lane & 15);               // 0..127
    float v = (kin < 32) ? Wx[(tap * 32 + kin) * 128 + cout]
                         : Wh[(tap * 32 + (kin - 32)) * 128 + cout];
    wfrag[idx] = f2bf(v);
}

// ---------------------------------------------------------------------------
// One recurrence step. Block = 16x16 spatial tile of one batch image,
// 256 threads = 8 waves; wave w computes tile rows {2w, 2w+1} x all 128 couts.
// ---------------------------------------------------------------------------
__global__ __launch_bounds__(256)
void convlstm_step_kernel(const float* __restrict__ x,        // (B,T,H,W,C)
                          const float* __restrict__ bias,     // (128)
                          const unsigned short* __restrict__ wfrag,
                          float* __restrict__ cstate,         // (B,H,W,F)
                          float* __restrict__ out,            // (B,T,1,H,W,F)
                          int t) {
    // 18x18 halo patch, 64 bf16 channels, channel-contiguous: 41472 B LDS.
    __shared__ __align__(16) unsigned short patch[PR * PR * CH];

    const int tid = threadIdx.x;
    const int bx = blockIdx.x, by = blockIdx.y, bb = blockIdx.z;

    // ---- stage patch: ch[0,32)=bf16(h+x), ch[32,64)=bf16(h); zero halo ----
    {
        const int grp = tid >> 3;          // pixel group (32 groups)
        const int sub = tid & 7;           // 4-channel sub-slot
        const float* xsl = x + ((size_t)bb * T_ + t) * (H_ * W_ * Cin);
        const float* hsl = (t > 0)
            ? out + ((size_t)bb * T_ + (t - 1)) * (H_ * W_ * Fo) : nullptr;
        for (int p = grp; p < PR * PR; p += 32) {   // uniform per wave
            int r  = p / PR;
            int cc = p - r * PR;
            int gy = by * TILE + r - 1;
            int gx = bx * TILE + cc - 1;
            float4 xv = make_float4(0.f, 0.f, 0.f, 0.f);
            float4 hv = make_float4(0.f, 0.f, 0.f, 0.f);
            if (gy >= 0 && gy < H_ && gx >= 0 && gx < W_) {
                size_t base = ((size_t)gy * W_ + gx) * Cin + sub * 4;
                xv = *(const float4*)(xsl + base);
                if (hsl) hv = *(const float4*)(hsl + base);
            }
            u16x4 av = { f2bf(hv.x + xv.x), f2bf(hv.y + xv.y),
                         f2bf(hv.z + xv.z), f2bf(hv.w + xv.w) };
            u16x4 hb = { f2bf(hv.x), f2bf(hv.y), f2bf(hv.z), f2bf(hv.w) };
            *(u16x4*)&patch[p * CH + sub * 4]      = av;
            *(u16x4*)&patch[p * CH + 32 + sub * 4] = hb;
        }
    }
    __syncthreads();

    // ---- implicit GEMM: M=16 pixels/row-tile, N=128, K=576 ----
    const int lane = tid & 31;
    const int wv   = tid >> 5;       // wave id 0..7 (uniform)
    const int g    = lane >> 4;      // half-wave group
    const int nn   = lane & 15;      // A: pixel-col M ; B/D: channel N

    f32x8 acc[2][NT];
    #pragma unroll
    for (int r = 0; r < 2; ++r)
        #pragma unroll
        for (int n = 0; n < NT; ++n)
            #pragma unroll
            for (int v = 0; v < 8; ++v) acc[r][n][v] = 0.0f;

    for (int kc = 0; kc < KC; ++kc) {
        const int tap = kc >> 1;
        const int dy  = tap / 3;
        const int dx  = tap - dy * 3;
        const int chb = ((kc & 1) << 5) + (g << 3);
        ABfrag a[2];
        #pragma unroll
        for (int r = 0; r < 2; ++r) {
            // A fragment: lane holds pixel (row 2w+r, col nn) of the tile;
            // two contiguous 16B LDS reads give K {0..7,16..23}+8g.
            int off = (((2 * wv + r) + dy) * PR + (nn + dx)) * CH + chb;
            a[r].h[0] = *(const bf16x8*)&patch[off];
            a[r].h[1] = *(const bf16x8*)&patch[off + 16];
        }
        #pragma unroll
        for (int n = 0; n < NT; ++n) {
            ABfrag bfr;  // 32B contiguous per lane, streamed via L0/L2
            size_t woff = (((size_t)kc * NT + n) * 32 + lane) * 16;
            bfr.h[0] = *(const bf16x8*)&wfrag[woff];
            bfr.h[1] = *(const bf16x8*)&wfrag[woff + 8];
            acc[0][n] = __builtin_amdgcn_wmma_f32_16x16x32_bf16(
                false, a[0].v, false, bfr.v, (short)0, acc[0][n], false, false);
            acc[1][n] = __builtin_amdgcn_wmma_f32_16x16x32_bf16(
                false, a[1].v, false, bfr.v, (short)0, acc[1][n], false, false);
        }
    }

    // ---- LSTM gate epilogue: i,f,c,o live in the same lane/slot ----
    float bv[NT];
    #pragma unroll
    for (int n = 0; n < NT; ++n) bv[n] = bias[(n << 4) + nn];

    #pragma unroll
    for (int r = 0; r < 2; ++r) {
        const int gy = by * TILE + 2 * wv + r;
        #pragma unroll
        for (int n0 = 0; n0 < 2; ++n0) {
            #pragma unroll
            for (int v = 0; v < 8; ++v) {
                // D layout: pixel-col M = v + 8g, channel N = nn.
                float zi = acc[r][n0    ][v] + bv[n0    ];
                float zf = acc[r][n0 + 2][v] + bv[n0 + 2];
                float zc = acc[r][n0 + 4][v] + bv[n0 + 4];
                float zo = acc[r][n0 + 6][v] + bv[n0 + 6];
                float ig = hsig(zi), fg = hsig(zf), og = hsig(zo);
                int gx = bx * TILE + v + (g << 3);
                int f0 = (n0 << 4) + nn;
                size_t ci = (((size_t)bb * H_ + gy) * W_ + gx) * Fo + f0;
                float cn = fg * cstate[ci] + ig * fast_tanh(zc);
                cstate[ci] = cn;
                out[(((size_t)bb * T_ + t) * (size_t)(H_ * W_)
                     + (size_t)gy * W_ + gx) * Fo + f0] = og * fast_tanh(cn);
            }
        }
    }
}

// ---------------------------------------------------------------------------
extern "C" void kernel_launch(void* const* d_in, const int* in_sizes, int n_in,
                              void* d_out, int out_size, void* d_ws, size_t ws_size,
                              hipStream_t stream) {
    const float* tensor = (const float*)d_in[0];   // (8,12,96,96,32)
    const float* Wx     = (const float*)d_in[1];   // (3,3,32,128)
    const float* Wh     = (const float*)d_in[2];   // (3,3,32,128)
    const float* bias   = (const float*)d_in[3];   // (128)
    float* out = (float*)d_out;                    // (8,12,1,96,96,32)

    unsigned short* wfrag = (unsigned short*)d_ws;                // 147456 B
    float* cstate = (float*)((char*)d_ws + 256 * 1024);           // 9.4 MB

    // c0 = 0 every call (deterministic); capture-legal async memset.
    hipMemsetAsync(cstate, 0, (size_t)NB * H_ * W_ * Fo * sizeof(float), stream);

    const int wtot = KC * NT * 32 * 16;
    prep_weights_kernel<<<(wtot + 255) / 256, 256, 0, stream>>>(Wx, Wh, wfrag);

    dim3 grid(W_ / TILE, H_ / TILE, NB);   // 6 x 6 x 8 = 288 blocks
    for (int t = 0; t < T_; ++t)
        convlstm_step_kernel<<<grid, 256, 0, stream>>>(tensor, bias, wfrag,
                                                       cstate, out, t);
}
